// MultiHeadSelfAttention_17952963297954
// MI455X (gfx1250) — matrix operand order repaired and checked by reading, not compiled, not run on hardware
//
#include <hip/hip_runtime.h>
#include <hip/hip_bf16.h>

// ---------------------------------------------------------------------------
// MultiHeadSelfAttention for MI455X (gfx1250, wave32, WMMA bf16)
//   B=4, S=2048, D=1024, H=16, Hd=64
// Pipeline:
//   1) qkv_gemm : Q/K/V = x @ W^T + b  (f32 in -> bf16 out, Q scaled by 1/8)
//   2) attn     : flash-attention; K tile staged via TDM (tensor_load_to_lds)
//   3) out_gemm : out = attnO @ Wo^T + bo ; A tile staged via async-to-LDS
// ---------------------------------------------------------------------------

#define DIM    1024
#define SEQ    2048
#define NB     4
#define NH     16
#define HD     64
#define MTOT   (NB * SEQ)   // 8192

typedef __attribute__((ext_vector_type(16))) __bf16 v16bf;
typedef __attribute__((ext_vector_type(8)))  __bf16 v8bf;
typedef __attribute__((ext_vector_type(2)))  __bf16 v2bf;
typedef __attribute__((ext_vector_type(8)))  float  v8f;

typedef __attribute__((ext_vector_type(4))) unsigned tdm_g0_t;
typedef __attribute__((ext_vector_type(8))) int      tdm_g1_t;
typedef __attribute__((ext_vector_type(4))) int      tdm_g2_t;
typedef __attribute__((ext_vector_type(8))) int      tdm_g4_t;

#if __has_builtin(__builtin_amdgcn_tensor_load_to_lds) && \
    __has_builtin(__builtin_amdgcn_s_wait_tensorcnt)
#define HAVE_TDM 1
#else
#define HAVE_TDM 0
#endif

static __device__ __forceinline__ v8f wmma_bf16(v16bf a, v16bf b, v8f c) {
    // D = A(16x32 bf16) * B(32x16 bf16) + C(16x16 f32)
    return __builtin_amdgcn_wmma_f32_16x16x32_bf16(
        false, a, false, b, (short)0, c, false, false);
}

// Build a 16-element bf16 fragment from two contiguous 16-byte chunks.
static __device__ __forceinline__ v16bf ld_frag(const __bf16* p0, const __bf16* p1) {
    union { v16bf v; v8bf h[2]; } u;
    u.h[0] = *(const v8bf*)p0;
    u.h[1] = *(const v8bf*)p1;
    return u.v;
}

// Packed f32x4 -> bf16x4 staging store (v_cvt_pk_bf16_f32 when available).
static __device__ __forceinline__ void store_bf16x4(__bf16* d, float4 v) {
#if __has_builtin(__builtin_amdgcn_cvt_pk_bf16_f32)
    *(v2bf*)(d)     = __builtin_amdgcn_cvt_pk_bf16_f32(v.x, v.y);
    *(v2bf*)(d + 2) = __builtin_amdgcn_cvt_pk_bf16_f32(v.z, v.w);
#else
    d[0] = (__bf16)v.x; d[1] = (__bf16)v.y; d[2] = (__bf16)v.z; d[3] = (__bf16)v.w;
#endif
}

// 32-bit LDS byte address of a shared-memory pointer (flat addr low bits).
static __device__ __forceinline__ unsigned lds_addr32(const void* p) {
    return (unsigned)(uintptr_t)p;
}

// Async global->LDS 16-byte copy (per-lane), tracked by ASYNCcnt.
static __device__ __forceinline__ void async_copy_b128(unsigned lds_off, const void* g) {
    asm volatile("global_load_async_to_lds_b128 %0, %1, off"
                 :: "v"(lds_off), "v"((unsigned long long)(uintptr_t)g)
                 : "memory");
}
static __device__ __forceinline__ void wait_async0() {
    asm volatile("s_wait_asynccnt 0" ::: "memory");
}

#if HAVE_TDM
// TDM: load a 2D bf16 tile [tile_h rows x tile_w elems] from global to LDS.
// LDS rows padded: +4 DWORDs (16 B) after every 32 DWORDs (128 B) ->
// effective LDS row stride = 144 B (matches KT_STRIDE below).
static __device__ __forceinline__ void tdm_load_k_tile(
    unsigned lds_off, const void* gtile, unsigned rows_left)
{
    const unsigned long long ga = (unsigned long long)(uintptr_t)gtile;
    tdm_g0_t g0;
    g0[0] = 1u;                                   // count=1, user mode
    g0[1] = lds_off;                              // lds_addr
    g0[2] = (unsigned)ga;                         // global_addr[31:0]
    g0[3] = (unsigned)((ga >> 32) & 0x1FFFFFFu)   // global_addr[56:32]
          | (2u << 30);                           // type=2 (image)
    tdm_g1_t g1;
    g1[0] = (int)((1u << 16)                      // data_size=1 -> 2 bytes
          | (1u << 20)                            // pad_enable
          | (4u << 22)                            // pad_interval: 32 DWORDs
          | (3u << 25));                          // pad_amount: 4 DWORDs
    g1[1] = (int)((unsigned)HD << 16);            // tensor_dim0 = 64 elems
    g1[2] = (int)(rows_left << 16);               // tensor_dim1 (rows in range)
    g1[3] = (int)((unsigned)HD << 16);            // tile_dim0 = 64 elems
    g1[4] = 32;                                   // tile_dim1 = 32 rows
    g1[5] = HD;                                   // tensor_dim0_stride = 64
    g1[6] = 0;
    g1[7] = 0;
    tdm_g2_t gz = {0, 0, 0, 0};
#if __clang_major__ >= 23
    tdm_g4_t g4 = {0, 0, 0, 0, 0, 0, 0, 0};
    __builtin_amdgcn_tensor_load_to_lds(g0, g1, gz, gz, g4, 0);
#else
    __builtin_amdgcn_tensor_load_to_lds(g0, g1, gz, gz, 0);
#endif
}
#endif

// LDS row strides (elements); byte strides are multiples of 16 for b128 loads.
#define AS_STRIDE 40   // 80 B
#define KT_STRIDE 72   // 144 B (= 128 B row + 16 B TDM pad)
#define VT_STRIDE 40   // 80 B

// ---------------------------------------------------------------------------
// Kernel 1: fused QKV projection.  grid = (8, 64, 3), block = 256 (8 waves)
// out layout: [B][H][S][Hd] bf16, Q pre-scaled by 1/sqrt(Hd)=0.125
// ---------------------------------------------------------------------------
__global__ __launch_bounds__(256) void qkv_gemm(
    const float* __restrict__ x,
    const float* __restrict__ Wq, const float* __restrict__ bq,
    const float* __restrict__ Wk, const float* __restrict__ bk,
    const float* __restrict__ Wv, const float* __restrict__ bvec,
    __bf16* __restrict__ Qb, __bf16* __restrict__ Kb, __bf16* __restrict__ Vb)
{
    const int z = blockIdx.z;
    const float* W    = (z == 0) ? Wq : ((z == 1) ? Wk : Wv);
    const float* bias = (z == 0) ? bq : ((z == 1) ? bk : bvec);
    __bf16* outp      = (z == 0) ? Qb : ((z == 1) ? Kb : Vb);
    const float scale = (z == 0) ? 0.125f : 1.0f;

    const int m0 = blockIdx.y * 128;
    const int n0 = blockIdx.x * 128;

    __shared__ __attribute__((aligned(16))) __bf16 As[128 * AS_STRIDE];
    __shared__ __attribute__((aligned(16))) __bf16 Bs[128 * AS_STRIDE];

    const int tid  = threadIdx.x;
    const int w    = tid >> 5;
    const int lane = tid & 31;
    const int ln   = lane & 15;
    const int half = lane >> 4;

    const int wm = (w & 3) * 32;   // 0,32,64,96
    const int wn = (w >> 2) * 64;  // 0,64

    v8f acc[2][4];
    #pragma unroll
    for (int i = 0; i < 2; ++i)
        #pragma unroll
        for (int j = 0; j < 4; ++j) acc[i][j] = (v8f){};

    const int lr = tid >> 3;        // 0..31
    const int lk = (tid & 7) * 4;   // 0..28

    for (int k0 = 0; k0 < DIM; k0 += 32) {
        __syncthreads();
        if (k0 + 32 < DIM) {  // warm L2/WGP$ for next K step
            __builtin_prefetch(x + (size_t)(m0 + lr) * DIM + k0 + 32 + lk, 0, 1);
            __builtin_prefetch(W + (size_t)(n0 + lr) * DIM + k0 + 32 + lk, 0, 1);
        }
        #pragma unroll
        for (int p = 0; p < 4; ++p) {
            const int r = lr + p * 32;
            const float4 v = *(const float4*)(x + (size_t)(m0 + r) * DIM + k0 + lk);
            store_bf16x4(&As[r * AS_STRIDE + lk], v);
        }
        #pragma unroll
        for (int p = 0; p < 4; ++p) {
            const int r = lr + p * 32;
            const float4 v = *(const float4*)(W + (size_t)(n0 + r) * DIM + k0 + lk);
            store_bf16x4(&Bs[r * AS_STRIDE + lk], v);
        }
        __syncthreads();

        v16bf af[2], bf[4];
        #pragma unroll
        for (int mt = 0; mt < 2; ++mt) {
            const __bf16* row = &As[(wm + mt * 16 + ln) * AS_STRIDE];
            af[mt] = ld_frag(row + half * 8, row + 16 + half * 8);
        }
        #pragma unroll
        for (int nt = 0; nt < 4; ++nt) {
            const __bf16* row = &Bs[(wn + nt * 16 + ln) * AS_STRIDE];
            bf[nt] = ld_frag(row + half * 8, row + 16 + half * 8);
        }
        #pragma unroll
        for (int mt = 0; mt < 2; ++mt)
            #pragma unroll
            for (int nt = 0; nt < 4; ++nt)
                acc[mt][nt] = wmma_bf16(af[mt], bf[nt], acc[mt][nt]);
    }

    // Epilogue: bias, scale, scatter to [B][H][S][Hd] bf16.
    #pragma unroll
    for (int nt = 0; nt < 4; ++nt) {
        const int n = n0 + wn + nt * 16 + ln;
        const float b = bias[n];
        const int h = n >> 6, hd = n & 63;
        #pragma unroll
        for (int mt = 0; mt < 2; ++mt) {
            #pragma unroll
            for (int r = 0; r < 8; ++r) {
                const int m = m0 + wm + mt * 16 + r + 8 * half;
                const int bb = m >> 11, s = m & (SEQ - 1);
                const float val = (acc[mt][nt][r] + b) * scale;
                outp[(((size_t)(bb * NH + h) * SEQ) + s) * HD + hd] = (__bf16)val;
            }
        }
    }
}

// ---------------------------------------------------------------------------
// Kernel 2: flash attention.  grid = (32 qtiles, 64 bh), block = 128 (4 waves)
// Each wave owns 16 queries x full Hd=64.  K-tiles of 32 keys, 64 iterations.
// K tile staged by the Tensor Data Mover; V tile staged transposed by VALU.
// ---------------------------------------------------------------------------
__global__ __launch_bounds__(128) void attn(
    const __bf16* __restrict__ Qb, const __bf16* __restrict__ Kb,
    const __bf16* __restrict__ Vb, __bf16* __restrict__ Ob)
{
    const int bh = blockIdx.y;
    const int q0 = blockIdx.x * 64;
    const int tid  = threadIdx.x;
    const int w    = tid >> 5;
    const int lane = tid & 31;
    const int ln   = lane & 15;
    const int half = lane >> 4;

    __shared__ __attribute__((aligned(16))) __bf16 Kt[32 * KT_STRIDE];
    __shared__ __attribute__((aligned(16))) __bf16 Vt[64 * VT_STRIDE];
    __shared__ __attribute__((aligned(16))) __bf16 Pt[4][16 * VT_STRIDE];

    // Q fragments (loaded once): A-layout, 16 queries x 64 hd (already *0.125)
    const __bf16* Qrow = Qb + ((size_t)bh * SEQ + q0 + w * 16 + ln) * HD;
    v16bf qf[2];
    qf[0] = ld_frag(Qrow + half * 8,      Qrow + 16 + half * 8);
    qf[1] = ld_frag(Qrow + 32 + half * 8, Qrow + 48 + half * 8);

    v8f acc[4];
    #pragma unroll
    for (int t = 0; t < 4; ++t) acc[t] = (v8f){};
    float mrow[8], rs[8];
    #pragma unroll
    for (int r = 0; r < 8; ++r) { mrow[r] = -1e30f; rs[r] = 0.0f; }

    for (int it = 0; it < SEQ / 32; ++it) {
        const int kt0 = it * 32;
        __syncthreads();

        // ---- Stage K tile [32 keys][64 hd] into Kt (144 B padded rows) ----
#if HAVE_TDM
        if (w == 0) {
            tdm_load_k_tile(lds_addr32(Kt),
                            Kb + ((size_t)bh * SEQ + kt0) * HD,
                            (unsigned)(SEQ - kt0));
            __builtin_amdgcn_s_wait_tensorcnt(0);
        }
#else
        #pragma unroll
        for (int p = 0; p < 2; ++p) {
            const int idx = tid + p * 128;
            const int key = idx >> 3, hc = (idx & 7) * 8;
            v8bf v = *(const v8bf*)(Kb + ((size_t)bh * SEQ + kt0 + key) * HD + hc);
            *(v8bf*)&Kt[key * KT_STRIDE + hc] = v;
        }
#endif
        // ---- Stage V tile transposed: Vt[hd][key] ----
        #pragma unroll
        for (int p = 0; p < 2; ++p) {
            const int idx = tid + p * 128;
            const int key = idx >> 3, hc = (idx & 7) * 8;
            v8bf v = *(const v8bf*)(Vb + ((size_t)bh * SEQ + kt0 + key) * HD + hc);
            #pragma unroll
            for (int e = 0; e < 8; ++e) Vt[(hc + e) * VT_STRIDE + key] = v[e];
        }
        if (it + 1 < SEQ / 32) {  // prefetch next V tile
            __builtin_prefetch(
                Vb + ((size_t)bh * SEQ + kt0 + 32 + (tid >> 3)) * HD + (tid & 7) * 8,
                0, 1);
        }
        __syncthreads();

        // Scores: 16 queries x 32 keys (two 16-key C tiles), K-dim = Hd = 64.
        v8f s[2];
        #pragma unroll
        for (int nt = 0; nt < 2; ++nt) {
            const __bf16* krow = &Kt[(nt * 16 + ln) * KT_STRIDE];
            v16bf b0 = ld_frag(krow + half * 8,      krow + 16 + half * 8);
            v16bf b1 = ld_frag(krow + 32 + half * 8, krow + 48 + half * 8);
            v8f zz = (v8f){};
            zz = wmma_bf16(qf[0], b0, zz);
            zz = wmma_bf16(qf[1], b1, zz);
            s[nt] = zz;
        }

        // Online softmax: rows live across 16-lane groups (wave32 C layout).
        #pragma unroll
        for (int r = 0; r < 8; ++r) {
            float mx = fmaxf(s[0][r], s[1][r]);
            #pragma unroll
            for (int d = 8; d >= 1; d >>= 1) mx = fmaxf(mx, __shfl_xor(mx, d, 16));
            const float mn = fmaxf(mrow[r], mx);
            const float alpha = __expf(mrow[r] - mn);
            mrow[r] = mn;
            const float p0 = __expf(s[0][r] - mn);
            const float p1 = __expf(s[1][r] - mn);
            rs[r] = rs[r] * alpha + p0 + p1;
            s[0][r] = p0; s[1][r] = p1;
            #pragma unroll
            for (int t = 0; t < 4; ++t) acc[t][r] *= alpha;
        }

        // Re-layout P: C-fragment -> LDS -> A-fragment (same-wave LDS in-order).
        #pragma unroll
        for (int nt = 0; nt < 2; ++nt)
            #pragma unroll
            for (int r = 0; r < 8; ++r)
                Pt[w][(r + 8 * half) * VT_STRIDE + nt * 16 + ln] = (__bf16)s[nt][r];
        asm volatile("" ::: "memory");

        const __bf16* prow = &Pt[w][ln * VT_STRIDE];
        const v16bf pf = ld_frag(prow + half * 8, prow + 16 + half * 8);
        #pragma unroll
        for (int t = 0; t < 4; ++t) {
            const __bf16* vrow = &Vt[(t * 16 + ln) * VT_STRIDE];
            v16bf bvf = ld_frag(vrow + half * 8, vrow + 16 + half * 8);
            acc[t] = wmma_bf16(pf, bvf, acc[t]);
        }
    }

    // Finalize: divide by row sum, store to [B][S][H*Hd] bf16.
    const int bb = bh >> 4, h = bh & 15;
    #pragma unroll
    for (int r = 0; r < 8; ++r) {
        float l = rs[r];
        #pragma unroll
        for (int d = 8; d >= 1; d >>= 1) l += __shfl_xor(l, d, 16);
        const float inv = 1.0f / l;
        const int q = q0 + w * 16 + r + 8 * half;
        #pragma unroll
        for (int t = 0; t < 4; ++t) {
            const int col = h * HD + t * 16 + ln;
            Ob[((size_t)bb * SEQ + q) * DIM + col] = (__bf16)(acc[t][r] * inv);
        }
    }
}

// ---------------------------------------------------------------------------
// Kernel 3: output projection (bf16 A via async-to-LDS, f32 W, f32 out).
// grid = (8, 64), block = 256
// ---------------------------------------------------------------------------
__global__ __launch_bounds__(256) void out_gemm(
    const __bf16* __restrict__ A, const float* __restrict__ W,
    const float* __restrict__ bias, float* __restrict__ outp)
{
    const int m0 = blockIdx.y * 128;
    const int n0 = blockIdx.x * 128;

    __shared__ __attribute__((aligned(16))) __bf16 As[128 * AS_STRIDE];
    __shared__ __attribute__((aligned(16))) __bf16 Bs[128 * AS_STRIDE];

    const int tid  = threadIdx.x;
    const int w    = tid >> 5;
    const int lane = tid & 31;
    const int ln   = lane & 15;
    const int half = lane >> 4;
    const int wm = (w & 3) * 32;
    const int wn = (w >> 2) * 64;

    v8f acc[2][4];
    #pragma unroll
    for (int i = 0; i < 2; ++i)
        #pragma unroll
        for (int j = 0; j < 4; ++j) acc[i][j] = (v8f){};

    const int ar = tid >> 2;        // 0..63
    const int ak = (tid & 3) * 8;   // 0,8,16,24
    const int lr = tid >> 3;        // 0..31
    const int lk = (tid & 7) * 4;   // 0..28

    for (int k0 = 0; k0 < DIM; k0 += 32) {
        __syncthreads();
        if (k0 + 32 < DIM) {
            __builtin_prefetch(W + (size_t)(n0 + lr) * DIM + k0 + 32 + lk, 0, 1);
        }
        // A tile: bf16, straight copy -> async global->LDS (ASYNCcnt path).
        #pragma unroll
        for (int p = 0; p < 2; ++p) {
            const int r = ar + p * 64;
            async_copy_b128(lds_addr32(&As[r * AS_STRIDE + ak]),
                            A + (size_t)(m0 + r) * DIM + k0 + ak);
        }
        // W tile: f32 -> bf16 conversion through VGPRs.
        #pragma unroll
        for (int p = 0; p < 4; ++p) {
            const int r = lr + p * 32;
            const float4 v = *(const float4*)(W + (size_t)(n0 + r) * DIM + k0 + lk);
            store_bf16x4(&Bs[r * AS_STRIDE + lk], v);
        }
        wait_async0();
        __syncthreads();

        v16bf af[2], bf[4];
        #pragma unroll
        for (int mt = 0; mt < 2; ++mt) {
            const __bf16* row = &As[(wm + mt * 16 + ln) * AS_STRIDE];
            af[mt] = ld_frag(row + half * 8, row + 16 + half * 8);
        }
        #pragma unroll
        for (int nt = 0; nt < 4; ++nt) {
            const __bf16* row = &Bs[(wn + nt * 16 + ln) * AS_STRIDE];
            bf[nt] = ld_frag(row + half * 8, row + 16 + half * 8);
        }
        #pragma unroll
        for (int mt = 0; mt < 2; ++mt)
            #pragma unroll
            for (int nt = 0; nt < 4; ++nt)
                acc[mt][nt] = wmma_bf16(af[mt], bf[nt], acc[mt][nt]);
    }

    #pragma unroll
    for (int nt = 0; nt < 4; ++nt) {
        const int n = n0 + wn + nt * 16 + ln;
        const float b = bias[n];
        #pragma unroll
        for (int mt = 0; mt < 2; ++mt) {
            #pragma unroll
            for (int r = 0; r < 8; ++r) {
                const int m = m0 + wm + mt * 16 + r + 8 * half;
                outp[(size_t)m * DIM + n] = acc[mt][nt][r] + b;
            }
        }
    }
}

// ---------------------------------------------------------------------------
extern "C" void kernel_launch(void* const* d_in, const int* in_sizes, int n_in,
                              void* d_out, int out_size, void* d_ws, size_t ws_size,
                              hipStream_t stream) {
    const float* x  = (const float*)d_in[0];
    const float* Wq = (const float*)d_in[1];
    const float* bq = (const float*)d_in[2];
    const float* Wk = (const float*)d_in[3];
    const float* bk = (const float*)d_in[4];
    const float* Wv = (const float*)d_in[5];
    const float* bv = (const float*)d_in[6];
    const float* Wo = (const float*)d_in[7];
    const float* bo = (const float*)d_in[8];
    float* out = (float*)d_out;

    const size_t qkv_elems = (size_t)NB * NH * SEQ * HD;  // 8,388,608
    char* ws = (char*)d_ws;
    __bf16* Qb = (__bf16*)(ws);
    __bf16* Kb = (__bf16*)(ws + 2 * qkv_elems);
    __bf16* Vb = (__bf16*)(ws + 4 * qkv_elems);
    __bf16* Ob = (__bf16*)(ws + 6 * qkv_elems);

    qkv_gemm<<<dim3(DIM / 128, MTOT / 128, 3), 256, 0, stream>>>(
        x, Wq, bq, Wk, bk, Wv, bv, Qb, Kb, Vb);
    attn<<<dim3(SEQ / 64, NB * NH), 128, 0, stream>>>(Qb, Kb, Vb, Ob);
    out_gemm<<<dim3(DIM / 128, MTOT / 128), 256, 0, stream>>>(Ob, Wo, bo, out);
}
